// NeRF_84945863180567
// MI455X (gfx1250) — matrix operand order repaired and checked by reading, not compiled, not run on hardware
//
#include <hip/hip_runtime.h>
#include <hip/hip_bf16.h>
#include <math.h>

// ---------------------------------------------------------------------------
// NeRF forward (coarse + fine) for gfx1250 using V_WMMA_F32_16X16X32_F16.
//
// Input order (setup_inputs dict, params pytrees flattened in sorted-key
// order: final(5x W,b), first(5x), second(4x), sigma(1x), third(1x)):
//   d_in[0]  ray (1024x6)
//   d_in[1]  u_c (1024x64)
//   d_in[2]  u_f (1024x128)
//   d_in[3..34]   params_c leaves (32)
//   d_in[35..66]  params_f leaves (32)
//   d_in[67] bg (3)
// d_out: [C_c (1024x3) | C_f (1024x3)] f32
// ---------------------------------------------------------------------------

typedef __attribute__((ext_vector_type(16))) _Float16 v16h;
typedef __attribute__((ext_vector_type(8)))  _Float16 v8h;
typedef __attribute__((ext_vector_type(8)))  float    v8f;

#define B_RAYS 1024
#define NC_    64
#define NF_    128
#define NS_F   (NC_ + NF_)
#define LX_    10
#define LD_    4
#define TN_    0.0f
#define TF_    2.5f
#define CHUNK  16384          // points per MLP chunk (divides 65536 and 196608)
#define KPAD_MAX 320          // largest padded K (second0: xe60|gap4|h256)

// ---------------------------------------------------------------------------
// Weight convert: f32 W[Kin x N] -> f16 transposed padded Wt[npad x kpad].
// Row remap handles the reference's concatenations as zero-filled gaps:
//   padded k < K1          -> source row k
//   K1 <= k < K1+gap       -> zero
//   k >= K1+gap            -> source row k-gap (if < Kin)
// ---------------------------------------------------------------------------
__global__ void conv_w_kernel(const float* __restrict__ W, int Kin, int N,
                              int K1, int gap, _Float16* __restrict__ Wt,
                              int kpad, int npad) {
  int idx = blockIdx.x * blockDim.x + threadIdx.x;
  if (idx >= kpad * npad) return;
  int k = idx % kpad;
  int n = idx / kpad;
  float v = 0.0f;
  int sr = (k < K1) ? k : ((k >= K1 + gap) ? (k - gap) : -1);
  if (n < N && sr >= 0 && sr < Kin) v = W[(size_t)sr * N + n];
  Wt[idx] = (_Float16)v;
}

// ---------------------------------------------------------------------------
// Core GEMM:  out = act( A[M x kpad](f16) @ Wt^T + bias )
//   A row-major (lda), Wt row-major [npad x kpad] (pre-transposed weights).
//
// Block = 128 threads (4 wave32s).  All 4 waves consume the SAME NT*16-column
// B tile, so it is staged once per block into LDS (<= 40 KB) and fragments
// are then read with ds_load_b128.  Each wave owns a distinct 16-row A tile;
// its A fragment is double-buffered from global (loaded one k-tile ahead);
// the final k-tile is peeled so the steady-state loop is branch-free.
//   grid.x = M/64, grid.y = npad/(NT*16)
//
// A fragment layout (this kernel's contiguous K order): lane = m + 16g holds
//   halves e=0..15 -> K = 16g + e          (contiguous v16h at +16g)
// B fragment: lane = n + 16g holds K = 16g .. 16g+15 (contiguous v16h of
// the transposed-weight row n, same K order as A).
// C fragment (f32 16x16): vgpr r, lane -> row = r + 8g, col = lane&15.
// act: 0 = none, 1 = relu, 2 = sigmoid
// ---------------------------------------------------------------------------
template <int NT>
__global__ __launch_bounds__(128)
void gemm_wmma(const _Float16* __restrict__ A, int lda,
               const _Float16* __restrict__ Wt, int kpad,
               const float* __restrict__ bias, int nreal,
               _Float16* __restrict__ outH, int ldo, int ocol,
               float* __restrict__ outF, int ldf, int act) {
  __shared__ _Float16 sB[64 * KPAD_MAX];

  const int tid  = threadIdx.x;
  const int wave = tid >> 5;
  const int lane = tid & 31;
  const int g    = lane >> 4;
  const int lq   = lane & 15;
  const int mrow = blockIdx.x * 64 + wave * 16;
  const int ncol = blockIdx.y * (NT * 16);

  // ---- stage the block's B tile (NT*16 rows of Wt) into LDS ----
  {
    const _Float16* src = Wt + (size_t)ncol * kpad;
    const int tileElems = NT * 16 * kpad;          // multiple of 8
    for (int i = tid * 8; i < tileElems; i += 128 * 8)
      *(v8h*)(sB + i) = *(const v8h*)(src + i);
  }
  __syncthreads();

  v8f acc[NT];
#pragma unroll
  for (int i = 0; i < NT; ++i)
#pragma unroll
    for (int r = 0; r < 8; ++r) acc[i][r] = 0.0f;

  const _Float16* pa = A + (size_t)(mrow + lq) * lda;
  const int ktc = kpad >> 5;

  auto loadA = [&](int kt) -> v16h {
    return *(const v16h*)(pa + kt * 32 + 16 * g);
  };
  auto ktile = [&](const v16h& a, int kt) {
#pragma unroll
    for (int nt = 0; nt < NT; ++nt) {
      v16h b = *(const v16h*)(sB + (nt * 16 + lq) * kpad + kt * 32 + 16 * g);
      acc[nt] = __builtin_amdgcn_wmma_f32_16x16x32_f16(
          false, a, false, b, (short)0, acc[nt], false, false);
    }
  };

  // steady state: always loading k-tile kt+1 while computing kt
  v16h a_cur = loadA(0);
  int kt = 0;
  for (; kt + 1 < ktc; ++kt) {
    v16h a_nxt = loadA(kt + 1);
    ktile(a_cur, kt);
    a_cur = a_nxt;
  }
  ktile(a_cur, kt);                                // peeled final k-tile

#pragma unroll
  for (int nt = 0; nt < NT; ++nt) {
    const int col = ncol + nt * 16 + lq;
    const float bb = (col < nreal) ? bias[col] : 0.0f;
#pragma unroll
    for (int r = 0; r < 8; ++r) {
      const int row = mrow + r + 8 * g;
      float v = acc[nt][r] + bb;
      if (act == 1) v = fmaxf(v, 0.0f);
      else if (act == 2) v = 1.0f / (1.0f + __expf(-v));
      if (outH) outH[(size_t)row * ldo + ocol + col] = (_Float16)v;
      if (outF && col < nreal) outF[(size_t)row * ldf + col] = v;
    }
  }
}

// ---------------------------------------------------------------------------
// t_c = TN + (j + u_c) * dt   (stratified coarse depths, sorted by design)
// ---------------------------------------------------------------------------
__global__ void tc_kernel(const float* __restrict__ u_c, float* __restrict__ tc) {
  int i = blockIdx.x * blockDim.x + threadIdx.x;
  if (i >= B_RAYS * NC_) return;
  const float dt = (TF_ - TN_) / (float)NC_;
  int j = i % NC_;
  tc[i] = TN_ + ((float)j + u_c[i]) * dt;
}

// ---------------------------------------------------------------------------
// Positional encoding of one chunk of points.  Writes:
//   x0  [CHUNK x 64]  : xe(60) | zeros(4)
//   cat [CHUNK x 320] : xe(60) | zeros(4) | (h written later at col 64)
//   fin [CHUNK x 288] : de(24) | zeros(8) | (v written later at col 32)
// ---------------------------------------------------------------------------
__global__ void encode_kernel(const float* __restrict__ ray,
                              const float* __restrict__ tarr, int p0, int ns,
                              _Float16* __restrict__ x0,
                              _Float16* __restrict__ cat,
                              _Float16* __restrict__ fin) {
  int q = blockIdx.x * blockDim.x + threadIdx.x;
  if (q >= CHUNK) return;
  int pq = p0 + q;
  int r  = pq / ns;
  float t  = tarr[pq];
  float o0 = ray[r * 6 + 0], o1 = ray[r * 6 + 1], o2 = ray[r * 6 + 2];
  float d0 = ray[r * 6 + 3], d1 = ray[r * 6 + 4], d2 = ray[r * 6 + 5];
  float xp[3] = {o0 + t * d0, o1 + t * d1, o2 + t * d2};
  float dp[3] = {d0, d1, d2};
  _Float16* px = x0 + (size_t)q * 64;
  _Float16* pc = cat + (size_t)q * 320;
  _Float16* pf = fin + (size_t)q * 288;
  const float PI_ = 3.14159265358979f;
  for (int c = 0; c < 3; ++c) {
    float p = tanhf(xp[c]);
    for (int l = 0; l < LX_; ++l) {
      float a = (float)(1 << l) * PI_ * p;
      float s, co;
      __sincosf(a, &s, &co);
      px[c * 2 * LX_ + l]       = (_Float16)s;
      px[c * 2 * LX_ + LX_ + l] = (_Float16)co;
      pc[c * 2 * LX_ + l]       = (_Float16)s;
      pc[c * 2 * LX_ + LX_ + l] = (_Float16)co;
    }
  }
  for (int j = 60; j < 64; ++j) { px[j] = (_Float16)0.0f; pc[j] = (_Float16)0.0f; }
  for (int c = 0; c < 3; ++c) {
    float p = tanhf(dp[c]);
    for (int l = 0; l < LD_; ++l) {
      float a = (float)(1 << l) * PI_ * p;
      float s, co;
      __sincosf(a, &s, &co);
      pf[c * 2 * LD_ + l]       = (_Float16)s;
      pf[c * 2 * LD_ + LD_ + l] = (_Float16)co;
    }
  }
  for (int j = 24; j < 32; ++j) pf[j] = (_Float16)0.0f;
}

// ---------------------------------------------------------------------------
// Volume rendering per ray: exclusive-cumsum transmittance.
// ---------------------------------------------------------------------------
__global__ void render_kernel(const float* __restrict__ sigma,
                              const float* __restrict__ rgb,
                              const float* __restrict__ tarr,
                              const float* __restrict__ bg, int ns,
                              float* __restrict__ Cout,
                              float* __restrict__ wout) {
  int r = blockIdx.x * blockDim.x + threadIdx.x;
  if (r >= B_RAYS) return;
  float macc = 0.0f, c0 = 0, c1 = 0, c2 = 0, wsum = 0;
  for (int j = 0; j < ns; ++j) {
    int q = r * ns + j;
    float delta = (j + 1 < ns) ? (tarr[q + 1] - tarr[q]) : 1.0e8f;
    float mass  = sigma[q] * delta;
    float Tj    = __expf(-macc);
    float alpha = 1.0f - __expf(-mass);
    float w     = Tj * alpha;
    macc += mass;
    c0 += w * rgb[q * 3 + 0];
    c1 += w * rgb[q * 3 + 1];
    c2 += w * rgb[q * 3 + 2];
    wsum += w;
    if (wout) wout[q] = w;
  }
  Cout[r * 3 + 0] = c0 + (1.0f - wsum) * bg[0];
  Cout[r * 3 + 1] = c1 + (1.0f - wsum) * bg[1];
  Cout[r * 3 + 2] = c2 + (1.0f - wsum) * bg[2];
}

// ---------------------------------------------------------------------------
// Inverse-CDF fine sampling + merge-sort with coarse depths (192 per ray).
// ---------------------------------------------------------------------------
__global__ void sample_fine_kernel(const float* __restrict__ wc,
                                   const float* __restrict__ u_f,
                                   const float* __restrict__ tc,
                                   float* __restrict__ tf) {
  int r = blockIdx.x * blockDim.x + threadIdx.x;
  if (r >= B_RAYS) return;
  float cdf[NC_ + 1];
  const float dt = (TF_ - TN_) / (float)NC_;
  float sum = 0.0f;
  for (int j = 0; j < NC_; ++j) sum += wc[r * NC_ + j] + 1e-5f;
  cdf[0] = 0.0f;
  float acc = 0.0f;
  for (int j = 0; j < NC_; ++j) {
    acc += (wc[r * NC_ + j] + 1e-5f) / sum;
    cdf[j + 1] = acc;
  }
  float vals[NS_F];
  for (int j = 0; j < NC_; ++j) vals[j] = tc[r * NC_ + j];
  for (int i = 0; i < NF_; ++i) {
    float u = u_f[r * NF_ + i];
    int lo = 0, hi = NC_ + 1;               // searchsorted(side='right')
    while (lo < hi) {
      int mid = (lo + hi) >> 1;
      if (cdf[mid] <= u) lo = mid + 1; else hi = mid;
    }
    int below = (lo - 1 < 0) ? 0 : lo - 1;
    int above = (lo < NC_) ? lo : NC_;
    float cg0 = cdf[below], cg1 = cdf[above];
    float pg0 = TN_ + below * dt, pg1 = TN_ + above * dt;
    float den = cg1 - cg0;
    if (den < 1e-5f) den = 1.0f;
    float tt = (u - cg0) / den;
    vals[NC_ + i] = pg0 + tt * (pg1 - pg0);
  }
  for (int i = 1; i < NS_F; ++i) {          // insertion sort (192)
    float v = vals[i];
    int j = i - 1;
    while (j >= 0 && vals[j] > v) { vals[j + 1] = vals[j]; --j; }
    vals[j + 1] = v;
  }
  for (int i = 0; i < NS_F; ++i) tf[r * NS_F + i] = vals[i];
}

// ---------------------------------------------------------------------------
// Host orchestration
// ---------------------------------------------------------------------------
struct LDesc { int lw; int Kin; int N; int K1; int gap; int kpad; int npad; };

// leaf offsets within one params block (sorted-key pytree flatten):
// final L0..L4 -> 0,2,4,6,8 ; first L0..L4 -> 10..18 ; second L0..L3 -> 20..26
// sigma -> 28 ; third -> 30  (bias = W leaf + 1)
static const LDesc gL[16] = {
    {10, 60, 256, 60, 0, 64, 256},   // 0  first0  (xe60 -> pad 64)
    {12, 256, 256, 256, 0, 256, 256},// 1  first1
    {14, 256, 256, 256, 0, 256, 256},// 2  first2
    {16, 256, 256, 256, 0, 256, 256},// 3  first3
    {18, 256, 256, 256, 0, 256, 256},// 4  first4
    {20, 316, 256, 60, 4, 320, 256}, // 5  second0 (xe60 | gap4 | h256)
    {22, 256, 256, 256, 0, 256, 256},// 6  second1
    {24, 256, 256, 256, 0, 256, 256},// 7  second2
    {26, 256, 256, 256, 0, 256, 256},// 8  second3
    {28, 256, 1, 256, 0, 256, 16},   // 9  sigma
    {30, 256, 256, 256, 0, 256, 256},// 10 third
    {0, 280, 128, 24, 8, 288, 128},  // 11 final0 (de24 | gap8 | v256)
    {2, 128, 128, 128, 0, 128, 128}, // 12 final1
    {4, 128, 128, 128, 0, 128, 128}, // 13 final2
    {6, 128, 128, 128, 0, 128, 128}, // 14 final3
    {8, 128, 3, 128, 0, 128, 16},    // 15 final4
};

extern "C" void kernel_launch(void* const* d_in, const int* in_sizes, int n_in,
                              void* d_out, int out_size, void* d_ws,
                              size_t ws_size, hipStream_t stream) {
  (void)in_sizes; (void)n_in; (void)out_size; (void)ws_size;

  const float* ray = (const float*)d_in[0];
  const float* u_c = (const float*)d_in[1];
  const float* u_f = (const float*)d_in[2];
  const float* bg  = (const float*)d_in[67];
  float* out = (float*)d_out;

  // ---- workspace bump allocator (256B aligned) ----
  char* wsb = (char*)d_ws;
  size_t off = 0;
  auto alloc = [&](size_t bytes) -> void* {
    void* p = wsb + off;
    off += (bytes + 255) & ~(size_t)255;
    return p;
  };

  // transposed f16 weights
  _Float16* wt[2][16];
  for (int net = 0; net < 2; ++net)
    for (int i = 0; i < 16; ++i)
      wt[net][i] = (_Float16*)alloc((size_t)gL[i].kpad * gL[i].npad * 2);

  const int NPF = B_RAYS * NS_F;                 // 196608 fine points
  float* tcbuf  = (float*)alloc((size_t)B_RAYS * NC_ * 4);
  float* tfbuf  = (float*)alloc((size_t)NPF * 4);
  float* sigbuf = (float*)alloc((size_t)NPF * 4);
  float* rgbbuf = (float*)alloc((size_t)NPF * 3 * 4);
  float* wcbuf  = (float*)alloc((size_t)B_RAYS * NC_ * 4);
  _Float16* x0  = (_Float16*)alloc((size_t)CHUNK * 64 * 2);
  _Float16* bfA = (_Float16*)alloc((size_t)CHUNK * 256 * 2);
  _Float16* bfB = (_Float16*)alloc((size_t)CHUNK * 256 * 2);
  _Float16* cat = (_Float16*)alloc((size_t)CHUNK * 320 * 2);
  _Float16* fin = (_Float16*)alloc((size_t)CHUNK * 288 * 2);
  _Float16* fa  = (_Float16*)alloc((size_t)CHUNK * 128 * 2);
  _Float16* fb  = (_Float16*)alloc((size_t)CHUNK * 128 * 2);

  // ---- weight conversion (both nets) ----
  for (int net = 0; net < 2; ++net) {
    int pbase = 3 + net * 32;
    for (int i = 0; i < 16; ++i) {
      int sz = gL[i].kpad * gL[i].npad;
      conv_w_kernel<<<(sz + 255) / 256, 256, 0, stream>>>(
          (const float*)d_in[pbase + gL[i].lw], gL[i].Kin, gL[i].N, gL[i].K1,
          gL[i].gap, wt[net][i], gL[i].kpad, gL[i].npad);
    }
  }

  const dim3 blk(128);
  auto run_chunk = [&](int net, int p0, const float* tarr, int ns) {
    int pbase = 3 + net * 32;
    auto bias = [&](int i) { return (const float*)d_in[pbase + gL[i].lw + 1]; };
    _Float16** W = wt[net];
    encode_kernel<<<CHUNK / 256, 256, 0, stream>>>(ray, tarr, p0, ns, x0, cat, fin);
    auto g4 = [&](int li, const _Float16* Ain, int lda, _Float16* oh, int ldo,
                  int ocol, int act) {
      dim3 grid(CHUNK / 64, gL[li].npad / 64);
      gemm_wmma<4><<<grid, blk, 0, stream>>>(Ain, lda, W[li], gL[li].kpad,
                                             bias(li), gL[li].N, oh, ldo, ocol,
                                             nullptr, 0, act);
    };
    // first stack (relu each)
    g4(0, x0, 64, bfA, 256, 0, 1);
    g4(1, bfA, 256, bfB, 256, 0, 1);
    g4(2, bfB, 256, bfA, 256, 0, 1);
    g4(3, bfA, 256, bfB, 256, 0, 1);
    g4(4, bfB, 256, cat, 320, 64, 1);   // h -> cat[:,64:320]
    // second stack: relu(out) on 0..2, raw h4 on 3
    g4(5, cat, 320, bfA, 256, 0, 1);
    g4(6, bfA, 256, bfB, 256, 0, 1);
    g4(7, bfB, 256, bfA, 256, 0, 1);
    g4(8, bfA, 256, bfB, 256, 0, 0);    // h4 (no relu)
    // sigma = relu(h4 @ Ws + bs) -> packed f32
    gemm_wmma<1><<<dim3(CHUNK / 64, 1), blk, 0, stream>>>(
        bfB, 256, W[9], gL[9].kpad, bias(9), 1, nullptr, 0, 0, sigbuf + p0, 1, 1);
    // v = relu(h4 @ Wt + bt) -> fin[:,32:288]
    g4(10, bfB, 256, fin, 288, 32, 1);
    // final stack
    g4(11, fin, 288, fa, 128, 0, 1);
    g4(12, fa, 128, fb, 128, 0, 1);
    g4(13, fb, 128, fa, 128, 0, 1);
    g4(14, fa, 128, fb, 128, 0, 1);
    // rgb = sigmoid(...) -> packed f32 [n,3]
    gemm_wmma<1><<<dim3(CHUNK / 64, 1), blk, 0, stream>>>(
        fb, 128, W[15], gL[15].kpad, bias(15), 3, nullptr, 0, 0,
        rgbbuf + (size_t)p0 * 3, 3, 2);
  };

  // ---- coarse pass ----
  tc_kernel<<<(B_RAYS * NC_ + 255) / 256, 256, 0, stream>>>(u_c, tcbuf);
  for (int c = 0; c < (B_RAYS * NC_) / CHUNK; ++c)
    run_chunk(0, c * CHUNK, tcbuf, NC_);
  render_kernel<<<(B_RAYS + 255) / 256, 256, 0, stream>>>(
      sigbuf, rgbbuf, tcbuf, bg, NC_, out, wcbuf);

  // ---- fine sampling + pass ----
  sample_fine_kernel<<<(B_RAYS + 255) / 256, 256, 0, stream>>>(wcbuf, u_f,
                                                               tcbuf, tfbuf);
  for (int c = 0; c < NPF / CHUNK; ++c)
    run_chunk(1, c * CHUNK, tfbuf, NS_F);
  render_kernel<<<(B_RAYS + 255) / 256, 256, 0, stream>>>(
      sigbuf, rgbbuf, tfbuf, bg, NS_F, out + (size_t)B_RAYS * 3, nullptr);
}
